// EEViT_PARAttention_60756607369646
// MI455X (gfx1250) — matrix operand order repaired and checked by reading, compiled, not run on hardware
//
#include <hip/hip_runtime.h>
#include <hip/hip_bf16.h>

// ---------------------------------------------------------------------------
// EEViT PAR Attention on MI455X (gfx1250): bf16 WMMA pipeline, f32 accumulate.
// B=8, N=1024, DIM=768, H=12, DH=64, LATENT=32, ctx=992.
// ---------------------------------------------------------------------------

typedef __bf16 bf16_t;
typedef __attribute__((ext_vector_type(16))) __bf16 v16bf;
typedef __attribute__((ext_vector_type(8)))  __bf16 v8bf;
typedef __attribute__((ext_vector_type(8)))  float  v8f;

#define WMMA_BF16(a, b, c) \
  __builtin_amdgcn_wmma_f32_16x16x32_bf16(false, (a), false, (b), (short)0, (c), false, false)

// Assemble a v16bf fragment from two contiguous 8-element (16-byte) chunks.
__device__ __forceinline__ v16bf load2x8(const bf16_t* p0, const bf16_t* p1) {
  union { v16bf v; v8bf h[2]; } u;
  u.h[0] = *(const v8bf*)p0;
  u.h[1] = *(const v8bf*)p1;
  return u.v;
}

// Pack two floats to 2x bf16 and store as one 32-bit LDS word.
__device__ __forceinline__ void store_pk(bf16_t* dst, float f0, float f1) {
  union { bf16_t h[2]; unsigned int u; } pk;
  pk.h[0] = (bf16_t)f0;
  pk.h[1] = (bf16_t)f1;
  *(unsigned int*)dst = pk.u;
}

// ---------------------------------------------------------------------------
// Kernel 1: q = x@Wq, kv = x@Wkv  (M=8192, K=768, N=768), bf16 outputs.
// Block tile 128x64, 8 waves (4x2), each wave 32x32 (2x2 WMMA accumulators).
// ---------------------------------------------------------------------------
__global__ __launch_bounds__(256)
void qkv_gemm_kernel(const float* __restrict__ x,
                     const float* __restrict__ Wq,
                     const float* __restrict__ Wkv,
                     bf16_t* __restrict__ qtmp,
                     bf16_t* __restrict__ kvtmp) {
  const float* __restrict__ W = (blockIdx.z == 0) ? Wq : Wkv;
  bf16_t* __restrict__ out    = (blockIdx.z == 0) ? qtmp : kvtmp;
  const int mblock = blockIdx.x * 128;
  const int nblock = blockIdx.y * 64;

  __shared__ __align__(16) bf16_t Alds[128][32];  // [m][k]
  __shared__ __align__(16) bf16_t Bt[64][32];     // transposed: [n][k]

  const int t    = threadIdx.x;
  const int lane = t & 31;
  const int wave = t >> 5;
  const int r0 = (wave >> 1) * 32;   // wave row origin
  const int c0 = (wave & 1) * 32;    // wave col origin

  v8f acc[2][2] = {};

  for (int kb = 0; kb < 768; kb += 32) {
    __syncthreads();
    // Stage A (x, f32 -> packed bf16): 128x32; 8 k-pairs/thread, b32 LDS stores.
    #pragma unroll
    for (int i = 0; i < 8; ++i) {
      int p = t + 256 * i;                 // 0..2047 pair index
      int k = (p & 15) * 2, m = p >> 4;
      const float* src = &x[(size_t)(mblock + m) * 768 + kb + k];
      store_pk(&Alds[m][k], src[0], src[1]);
    }
    // Stage B transposed (W, f32 -> bf16): 32x64 -> Bt[n][k]; k-pairs, b32 stores.
    #pragma unroll
    for (int i = 0; i < 4; ++i) {
      int p = t + 256 * i;                 // 0..1023 pair index
      int n = p & 63, k = (p >> 6) * 2;
      float w0 = W[(size_t)(kb + k) * 768 + nblock + n];
      float w1 = W[(size_t)(kb + k + 1) * 768 + nblock + n];
      store_pk(&Bt[n][k], w0, w1);
    }
    // Prefetch next K tile into cache while WMMAs run.
    if (kb + 32 < 768) {
      __builtin_prefetch(&x[(size_t)(mblock + (t >> 1)) * 768 + kb + 32 + (t & 1) * 16], 0, 1);
      __builtin_prefetch(&W[(size_t)(kb + 32 + (t >> 3)) * 768 + nblock + (t & 7) * 8], 0, 1);
    }
    __syncthreads();

    const int mA   = lane & 15;
    const int koff = (lane >> 4) * 8;    // unused; A frag K offset is kB too
    const int kB   = (lane >> 4) * 16;   // B frag K offset per lane half
    v16bf a0 = load2x8(&Alds[r0 + mA][kB],      &Alds[r0 + mA][kB + 8]);
    v16bf a1 = load2x8(&Alds[r0 + 16 + mA][kB], &Alds[r0 + 16 + mA][kB + 8]);
    v16bf b0 = load2x8(&Bt[c0 + mA][kB],          &Bt[c0 + mA][kB + 8]);
    v16bf b1 = load2x8(&Bt[c0 + 16 + mA][kB],     &Bt[c0 + 16 + mA][kB + 8]);

    acc[0][0] = WMMA_BF16(a0, b0, acc[0][0]);
    acc[0][1] = WMMA_BF16(a0, b1, acc[0][1]);
    acc[1][0] = WMMA_BF16(a1, b0, acc[1][0]);
    acc[1][1] = WMMA_BF16(a1, b1, acc[1][1]);
  }

  // C layout: N = lane&15, M = r + (lane>=16)*8
  const int nC = lane & 15;
  const int rh = (lane >> 4) * 8;
  #pragma unroll
  for (int i = 0; i < 2; ++i)
    #pragma unroll
    for (int j = 0; j < 2; ++j)
      #pragma unroll
      for (int r = 0; r < 8; ++r) {
        int m = mblock + r0 + i * 16 + rh + r;
        int n = nblock + c0 + j * 16 + nC;
        out[(size_t)m * 768 + n] = (bf16_t)acc[i][j][r];
      }
}

// ---------------------------------------------------------------------------
// Kernel 2: RoPE (+ q scale) and LayerNorm, reshape to head-major layout.
// Block = one (b,n) position, 12 waves = 12 heads; lane owns pair (dh, dh+32).
// ---------------------------------------------------------------------------
__global__ __launch_bounds__(384)
void rope_ln_kernel(const bf16_t* __restrict__ qtmp,
                    const bf16_t* __restrict__ kvtmp,
                    const float* __restrict__ ln_g,
                    const float* __restrict__ ln_b,
                    bf16_t* __restrict__ qh,
                    bf16_t* __restrict__ kvh,
                    bf16_t* __restrict__ lkvh) {
  const int row  = blockIdx.x;        // 0..8191 == b*1024 + n
  const int b    = row >> 10;
  const int n    = row & 1023;
  const int h    = threadIdx.x >> 5;  // head
  const int lane = threadIdx.x & 31;  // dh pair (lane, lane+32)

  const size_t base_in = (size_t)row * 768 + h * 64;
  float q1 = (float)qtmp[base_in + lane];
  float q2 = (float)qtmp[base_in + 32 + lane];
  float k1 = (float)kvtmp[base_in + lane];
  float k2 = (float)kvtmp[base_in + 32 + lane];

  float inv = __powf(10000.f, -(float)lane * (1.f / 32.f));
  float f = (float)n * inv;
  float c = __cosf(f), s = __sinf(f);

  float qo1 = (q1 * c - q2 * s) * 0.125f;   // DH^-0.5 = 1/8
  float qo2 = (q2 * c + q1 * s) * 0.125f;
  float ko1 = k1 * c - k2 * s;
  float ko2 = k2 * c + k1 * s;

  // LayerNorm over 64 values (2 per lane) via wave32 shuffles
  float sum = ko1 + ko2;
  float sq  = ko1 * ko1 + ko2 * ko2;
  #pragma unroll
  for (int m = 16; m >= 1; m >>= 1) {
    sum += __shfl_xor(sum, m, 32);
    sq  += __shfl_xor(sq,  m, 32);
  }
  float mu   = sum * (1.f / 64.f);
  float var  = sq * (1.f / 64.f) - mu * mu;
  float rstd = __frsqrt_rn(var + 1e-5f);
  float l1 = (ko1 - mu) * rstd * ln_g[lane]      + ln_b[lane];
  float l2 = (ko2 - mu) * rstd * ln_g[lane + 32] + ln_b[lane + 32];

  const size_t base_out = ((size_t)(b * 12 + h) * 1024 + n) * 64;
  qh[base_out + lane]        = (bf16_t)qo1;
  qh[base_out + 32 + lane]   = (bf16_t)qo2;
  kvh[base_out + lane]       = (bf16_t)ko1;
  kvh[base_out + 32 + lane]  = (bf16_t)ko2;
  lkvh[base_out + lane]      = (bf16_t)l1;
  lkvh[base_out + 32 + lane] = (bf16_t)l2;
}

// ---------------------------------------------------------------------------
// Kernel 3: flash-style attention. Block = (bh, 32-query block), 2 waves,
// each wave owns 16 query rows; keys processed 32 at a time. Klen is 992 for
// context queries, 1024 for the last (latent) 32-row block.
// ---------------------------------------------------------------------------
__global__ __launch_bounds__(64)
void attn_kernel(const bf16_t* __restrict__ qh,
                 const bf16_t* __restrict__ kvh,
                 const bf16_t* __restrict__ lkvh,
                 bf16_t* __restrict__ merged) {
  const int bh   = blockIdx.x;       // 0..95 = b*12 + h
  const int qblk = blockIdx.y;       // 0..31
  const int tid  = threadIdx.x;
  const int wave = tid >> 5;
  const int lane = tid & 31;
  const int qrow0 = qblk * 32 + wave * 16;
  const int Klen  = (qblk == 31) ? 1024 : 992;

  __shared__ __align__(16) bf16_t Vt[64][32];        // transposed V: [dh][key]
  __shared__ __align__(16) bf16_t Plds[2][16][32];   // per-wave P staging

  const bf16_t* __restrict__ qbase = qh   + (size_t)bh * 1024 * 64;
  const bf16_t* __restrict__ kbase = kvh  + (size_t)bh * 1024 * 64;
  const bf16_t* __restrict__ vbase = lkvh + (size_t)bh * 1024 * 64;

  const int mA   = lane & 15;
  const int koff = (lane >> 4) * 8;
  const int kB   = (lane >> 4) * 16;

  // q A-fragments (DH=64 -> two 16x32 frags), contiguous 16B loads per lane
  const bf16_t* qp = qbase + (size_t)(qrow0 + mA) * 64;
  v16bf aq0 = load2x8(qp + kB,      qp + kB + 8);
  v16bf aq1 = load2x8(qp + 32 + kB, qp + 32 + kB + 8);

  float mrow[8], lrow[8];
  #pragma unroll
  for (int r = 0; r < 8; ++r) { mrow[r] = -1e30f; lrow[r] = 0.f; }
  v8f acc[4] = {};

  for (int kb = 0; kb < Klen; kb += 32) {
    __syncthreads();
    // Stage V chunk transposed: 32 keys x 64 dh -> Vt[dh][key].
    // Pack key-pairs: one b32 LDS store per pair, global loads coalesced in dh.
    #pragma unroll
    for (int i = 0; i < 16; ++i) {
      int p = tid + 64 * i;              // 0..1023 pair index
      int dh = p & 63, key = (p >> 6) * 2;
      float v0 = (float)vbase[(size_t)(kb + key) * 64 + dh];
      float v1 = (float)vbase[(size_t)(kb + key + 1) * 64 + dh];
      store_pk(&Vt[dh][key], v0, v1);
    }
    // Prefetch next key chunk (K and V rows) while this chunk computes.
    if (kb + 32 < Klen) {
      __builtin_prefetch(&kbase[(size_t)(kb + 32 + (tid >> 1)) * 64 + (tid & 1) * 32], 0, 1);
      __builtin_prefetch(&vbase[(size_t)(kb + 32 + (tid >> 1)) * 64 + (tid & 1) * 32], 0, 1);
    }
    __syncthreads();

    // Scores S = q @ kv^T for two 16-key subtiles (K-dim = DH = 64 -> 2 WMMA)
    v8f S[2];
    #pragma unroll
    for (int s2 = 0; s2 < 2; ++s2) {
      const bf16_t* kp = kbase + (size_t)(kb + s2 * 16 + mA) * 64;
      v16bf b0 = load2x8(kp + kB,      kp + kB + 8);       // dh 0..31
      v16bf b1 = load2x8(kp + 32 + kB, kp + 32 + kB + 8);  // dh 32..63
      v8f sacc = {};
      sacc = WMMA_BF16(aq0, b0, sacc);
      sacc = WMMA_BF16(aq1, b1, sacc);
      S[s2] = sacc;
    }

    // Online softmax per row (row r + (lane>=16)*8 lives across a 16-lane half)
    #pragma unroll
    for (int r = 0; r < 8; ++r) {
      float v = fmaxf(S[0][r], S[1][r]);
      #pragma unroll
      for (int m2 = 8; m2 >= 1; m2 >>= 1) v = fmaxf(v, __shfl_xor(v, m2, 32));
      float mnew  = fmaxf(mrow[r], v);
      float scale = __expf(mrow[r] - mnew);
      mrow[r] = mnew;
      float p0 = __expf(S[0][r] - mnew);
      float p1 = __expf(S[1][r] - mnew);
      float rs = p0 + p1;
      #pragma unroll
      for (int m2 = 8; m2 >= 1; m2 >>= 1) rs += __shfl_xor(rs, m2, 32);
      lrow[r] = lrow[r] * scale + rs;
      #pragma unroll
      for (int tn = 0; tn < 4; ++tn) acc[tn][r] *= scale;
      // C-layout -> LDS (re-layout P into A-fragment form)
      int m = r + (lane >> 4) * 8;
      Plds[wave][m][mA]      = (bf16_t)p0;
      Plds[wave][m][mA + 16] = (bf16_t)p1;
    }
    __syncthreads();

    // acc += P @ V  (K = 32 keys, 4 output dh-tiles of 16)
    v16bf pa = load2x8(&Plds[wave][mA][kB], &Plds[wave][mA][kB + 8]);
    #pragma unroll
    for (int tn = 0; tn < 4; ++tn) {
      v16bf vb = load2x8(&Vt[tn * 16 + mA][kB], &Vt[tn * 16 + mA][kB + 8]);
      acc[tn] = WMMA_BF16(pa, vb, acc[tn]);
    }
  }

  // Epilogue: normalize and write merged (B, N, DIM) in bf16
  const int b = bh / 12, h = bh % 12;
  #pragma unroll
  for (int r = 0; r < 8; ++r) {
    float invl = 1.f / lrow[r];
    int m = qrow0 + r + (lane >> 4) * 8;
    size_t rowbase = ((size_t)b * 1024 + m) * 768 + h * 64;
    #pragma unroll
    for (int tn = 0; tn < 4; ++tn)
      merged[rowbase + tn * 16 + mA] = (bf16_t)(acc[tn][r] * invl);
  }
}

// ---------------------------------------------------------------------------
// Kernel 4: out = merged @ (Wo_ctx | Wo_lat) + bias, f32 output.
// Block tile 32x64 (4 waves, 16x32 each); 992 boundary aligns with 32-row
// tiles, so each block picks one weight/bias pair.
// ---------------------------------------------------------------------------
__global__ __launch_bounds__(128)
void out_proj_kernel(const bf16_t* __restrict__ merged,
                     const float* __restrict__ Wo_ctx,
                     const float* __restrict__ bo_ctx,
                     const float* __restrict__ Wo_lat,
                     const float* __restrict__ bo_lat,
                     float* __restrict__ out) {
  const int mblock = blockIdx.x * 32;
  const int nblock = blockIdx.y * 64;
  const bool lat = ((mblock & 1023) >= 992);
  const float* __restrict__ W    = lat ? Wo_lat : Wo_ctx;
  const float* __restrict__ bias = lat ? bo_lat : bo_ctx;

  __shared__ __align__(16) bf16_t Alds[32][32];
  __shared__ __align__(16) bf16_t Bt[64][32];

  const int t    = threadIdx.x;
  const int lane = t & 31;
  const int wave = t >> 5;
  const int r0 = (wave >> 1) * 16;
  const int c0 = (wave & 1) * 32;

  v8f acc[2] = {};

  for (int kb = 0; kb < 768; kb += 32) {
    __syncthreads();
    // Stage A (bf16 copy): 32x32; k-pair b32 copies.
    #pragma unroll
    for (int i = 0; i < 4; ++i) {
      int p = t + 128 * i;               // 0..511 pair index
      int k = (p & 15) * 2, m = p >> 4;
      *(unsigned int*)&Alds[m][k] =
          *(const unsigned int*)&merged[(size_t)(mblock + m) * 768 + kb + k];
    }
    // Stage B transposed (f32 -> bf16), k-pair b32 stores.
    #pragma unroll
    for (int i = 0; i < 8; ++i) {
      int p = t + 128 * i;               // 0..1023 pair index
      int n = p & 63, k = (p >> 6) * 2;
      float w0 = W[(size_t)(kb + k) * 768 + nblock + n];
      float w1 = W[(size_t)(kb + k + 1) * 768 + nblock + n];
      store_pk(&Bt[n][k], w0, w1);
    }
    if (kb + 32 < 768) {
      __builtin_prefetch(&merged[(size_t)(mblock + (t >> 2)) * 768 + kb + 32 + (t & 3) * 8], 0, 1);
      __builtin_prefetch(&W[(size_t)(kb + 32 + (t >> 2)) * 768 + nblock + (t & 3) * 16], 0, 1);
    }
    __syncthreads();

    const int mA   = lane & 15;
    const int koff = (lane >> 4) * 8;
    const int kB   = (lane >> 4) * 16;
    v16bf a  = load2x8(&Alds[r0 + mA][kB],  &Alds[r0 + mA][kB + 8]);
    v16bf b0 = load2x8(&Bt[c0 + mA][kB],      &Bt[c0 + mA][kB + 8]);
    v16bf b1 = load2x8(&Bt[c0 + 16 + mA][kB], &Bt[c0 + 16 + mA][kB + 8]);
    acc[0] = WMMA_BF16(a, b0, acc[0]);
    acc[1] = WMMA_BF16(a, b1, acc[1]);
  }

  const int nC = lane & 15;
  const int rh = (lane >> 4) * 8;
  #pragma unroll
  for (int j = 0; j < 2; ++j)
    #pragma unroll
    for (int r = 0; r < 8; ++r) {
      int m = mblock + r0 + rh + r;
      int n = nblock + c0 + j * 16 + nC;
      out[(size_t)m * 768 + n] = acc[j][r] + bias[n];
    }
}

// ---------------------------------------------------------------------------
// Launch: inputs are {x, Wq, Wkv, Wo_ctx, bo_ctx, Wo_lat, bo_lat, ln_g, ln_b}
// ---------------------------------------------------------------------------
extern "C" void kernel_launch(void* const* d_in, const int* in_sizes, int n_in,
                              void* d_out, int out_size, void* d_ws, size_t ws_size,
                              hipStream_t stream) {
  const float* x      = (const float*)d_in[0];
  const float* Wq     = (const float*)d_in[1];
  const float* Wkv    = (const float*)d_in[2];
  const float* Wo_ctx = (const float*)d_in[3];
  const float* bo_ctx = (const float*)d_in[4];
  const float* Wo_lat = (const float*)d_in[5];
  const float* bo_lat = (const float*)d_in[6];
  const float* ln_g   = (const float*)d_in[7];
  const float* ln_b   = (const float*)d_in[8];
  float* out = (float*)d_out;

  const size_t NE = (size_t)8192 * 768;   // elements per (B*N, DIM) plane
  bf16_t* qtmp   = (bf16_t*)d_ws;
  bf16_t* kvtmp  = qtmp + NE;
  bf16_t* qh     = kvtmp + NE;
  bf16_t* kvh    = qh + NE;
  bf16_t* lkvh   = kvh + NE;
  bf16_t* merged = lkvh + NE;

  dim3 g1(64, 12, 2);                      // 8192/128 x 768/64 x {Wq,Wkv}
  qkv_gemm_kernel<<<g1, 256, 0, stream>>>(x, Wq, Wkv, qtmp, kvtmp);

  rope_ln_kernel<<<8192, 384, 0, stream>>>(qtmp, kvtmp, ln_g, ln_b, qh, kvh, lkvh);

  dim3 g3(96, 32);                         // (B*H) x (1024/32 query blocks)
  attn_kernel<<<g3, 64, 0, stream>>>(qh, kvh, lkvh, merged);

  dim3 g4(256, 12);                        // 8192/32 x 768/64
  out_proj_kernel<<<g4, 128, 0, stream>>>(merged, Wo_ctx, bo_ctx, Wo_lat, bo_lat, out);
}
